// BitLinearNew_74139725463731
// MI455X (gfx1250) — 8-bit matrix operand order repaired and checked by reading, compiled, not run on hardware
//
#include <hip/hip_runtime.h>

typedef __attribute__((ext_vector_type(8))) int v8i;

#define TOKENS   8192
#define KDIM     4096
#define NDIM     4096
#define TILE_M   128
#define TILE_N   128
#define TILE_K   64
#define LDSROW   80
#define KTILES   (KDIM / TILE_K)
#define ABUF     (TILE_M * LDSROW)
#define BBUF     (TILE_N * LDSROW)

__device__ __forceinline__ void async_b128(unsigned lds_addr, unsigned long long gva) {
    asm volatile("global_load_async_to_lds_b128 %0, %1, off"
                 :: "v"(lds_addr), "v"(gva) : "memory");
}

__device__ __forceinline__ signed char quant1(float f, float qs) {
    int q = __float2int_rn(f * qs);
    q = q > 127 ? 127 : (q < -127 ? -127 : q);
    return (signed char)q;
}

__global__ __launch_bounds__(256) void bitnet_quant_x(
    const float* __restrict__ x, const float* __restrict__ scale,
    signed char* __restrict__ xq, float* __restrict__ gscale)
{
    __shared__ float red[256];
    const int tok = blockIdx.x;
    const int t   = threadIdx.x;
    const float4* xr = (const float4*)(x + (size_t)tok * KDIM);

    float4 v[4];
    float m = 0.0f;
#pragma unroll
    for (int i = 0; i < 4; ++i) {
        v[i] = xr[t + 256 * i];
        m = fmaxf(m, fmaxf(fmaxf(fabsf(v[i].x), fabsf(v[i].y)),
                           fmaxf(fabsf(v[i].z), fabsf(v[i].w))));
    }
    red[t] = m;
    __syncthreads();
    for (int s = 128; s > 0; s >>= 1) {
        if (t < s) red[t] = fmaxf(red[t], red[t + s]);
        __syncthreads();
    }
    const float gamma = fmaxf(red[0], 1e-5f);
    const float qs = 127.0f / gamma;
    if (t == 0) gscale[tok] = gamma * (1.0f / 127.0f) * scale[0];

    char4* xqr = (char4*)(xq + (size_t)tok * KDIM);
#pragma unroll
    for (int i = 0; i < 4; ++i) {
        char4 q;
        q.x = quant1(v[i].x, qs);
        q.y = quant1(v[i].y, qs);
        q.z = quant1(v[i].z, qs);
        q.w = quant1(v[i].w, qs);
        xqr[t + 256 * i] = q;
    }
}

__global__ __launch_bounds__(256) void bitnet_quant_w(
    const float* __restrict__ w, signed char* __restrict__ wq)
{
    const size_t idx = (size_t)blockIdx.x * 256 + threadIdx.x;
    const float4 f = ((const float4*)w)[idx];
    char4 q;
    q.x = (signed char)__float2int_rn(f.x);
    q.y = (signed char)__float2int_rn(f.y);
    q.z = (signed char)__float2int_rn(f.z);
    q.w = (signed char)__float2int_rn(f.w);
    ((char4*)wq)[idx] = q;
}

__global__ __launch_bounds__(256) void bitnet_gemm_iu8(
    const signed char* __restrict__ xq, const signed char* __restrict__ wq,
    const float* __restrict__ gscale, float* __restrict__ out)
{
    __shared__ signed char lA[2][ABUF];
    __shared__ signed char lB[2][BBUF];
    __shared__ float gs[TILE_M];

    const int bm   = blockIdx.x;
    const int bn   = blockIdx.y;
    const int t    = threadIdx.x;
    const int lane = t & 31;
    const int wv   = t >> 5;
    const int wm   = (wv & 3) * 32;
    const int wn   = (wv >> 2) * 64;
    const int hf   = lane >> 4;
    const int lm   = lane & 15;

    if (t < TILE_M) gs[t] = gscale[bm * TILE_M + t];

    const int row0 = t >> 2,         seg0 = (t & 3) * 16;
    const int row1 = (t + 256) >> 2, seg1 = ((t + 256) & 3) * 16;

    const unsigned ldsA0 = (unsigned)(uintptr_t)&lA[0][row0 * LDSROW + seg0];
    const unsigned ldsA1 = (unsigned)(uintptr_t)&lA[0][row1 * LDSROW + seg1];
    const unsigned ldsB0 = (unsigned)(uintptr_t)&lB[0][row0 * LDSROW + seg0];
    const unsigned ldsB1 = (unsigned)(uintptr_t)&lB[0][row1 * LDSROW + seg1];

    const signed char* gA0 = xq + (size_t)(bm * TILE_M + row0) * KDIM + seg0;
    const signed char* gA1 = xq + (size_t)(bm * TILE_M + row1) * KDIM + seg1;
    const signed char* gB0 = wq + (size_t)(bn * TILE_N + row0) * KDIM + seg0;
    const signed char* gB1 = wq + (size_t)(bn * TILE_N + row1) * KDIM + seg1;

    auto issue_slice = [&](int buf, int kt) {
        const unsigned ao = (unsigned)(buf * ABUF);
        const unsigned bo = (unsigned)(buf * BBUF);
        const size_t   ko = (size_t)kt * TILE_K;
        async_b128(ldsA0 + ao, (unsigned long long)(uintptr_t)(gA0 + ko));
        async_b128(ldsA1 + ao, (unsigned long long)(uintptr_t)(gA1 + ko));
        async_b128(ldsB0 + bo, (unsigned long long)(uintptr_t)(gB0 + ko));
        async_b128(ldsB1 + bo, (unsigned long long)(uintptr_t)(gB1 + ko));
    };

    v8i acc[2][4] = {};

    auto compute_slice = [&](int buf) {
        v8i afrag[2];
#pragma unroll
        for (int f = 0; f < 2; ++f) {
            const signed char* ap = &lA[buf][(wm + f * 16 + lm) * LDSROW + 8 * hf];
            uint2 p0 = *(const uint2*)(ap);
            uint2 p1 = *(const uint2*)(ap + 16);
            uint2 p2 = *(const uint2*)(ap + 32);
            uint2 p3 = *(const uint2*)(ap + 48);
            afrag[f] = (v8i){(int)p0.x, (int)p0.y, (int)p1.x, (int)p1.y,
                             (int)p2.x, (int)p2.y, (int)p3.x, (int)p3.y};
        }

#pragma unroll
        for (int g = 0; g < 4; ++g) {
            const signed char* bp = &lB[buf][(wn + g * 16 + lm) * LDSROW + 8 * hf];
            uint2 q0 = *(const uint2*)(bp);
            uint2 q1 = *(const uint2*)(bp + 16);
            uint2 q2 = *(const uint2*)(bp + 32);
            uint2 q3 = *(const uint2*)(bp + 48);
            v8i bfrag = (v8i){(int)q0.x, (int)q0.y, (int)q1.x, (int)q1.y,
                              (int)q2.x, (int)q2.y, (int)q3.x, (int)q3.y};
#pragma unroll
            for (int f = 0; f < 2; ++f) {
                acc[f][g] = __builtin_amdgcn_wmma_i32_16x16x64_iu8(
                    true, afrag[f], true, bfrag, acc[f][g], false, false);
            }
        }
    };

    issue_slice(0, 0);

#pragma unroll 1
    for (int kt = 0; kt < KTILES - 1; ++kt) {
        const int buf = kt & 1;
        issue_slice(buf ^ 1, kt + 1);
        asm volatile("s_wait_asynccnt 0x4" ::: "memory");
        __syncthreads();
        compute_slice(buf);
        __syncthreads();
    }

    asm volatile("s_wait_asynccnt 0x0" ::: "memory");
    __syncthreads();
    compute_slice((KTILES - 1) & 1);

#pragma unroll
    for (int f = 0; f < 2; ++f) {
#pragma unroll
        for (int g = 0; g < 4; ++g) {
#pragma unroll
            for (int r = 0; r < 8; ++r) {
                const int ml = wm + f * 16 + 8 * hf + r;
                const int nl = wn + g * 16 + lm;
                const size_t o = (size_t)(bm * TILE_M + ml) * NDIM + (bn * TILE_N + nl);
                out[o] = (float)acc[f][g][r] * gs[ml];
            }
        }
    }
}

extern "C" void kernel_launch(void* const* d_in, const int* in_sizes, int n_in,
                              void* d_out, int out_size, void* d_ws, size_t ws_size,
                              hipStream_t stream) {
    const float* x     = (const float*)d_in[0];
    const float* w     = (const float*)d_in[1];
    const float* scale = (const float*)d_in[2];
    float* out = (float*)d_out;

    signed char* xq = (signed char*)d_ws;
    signed char* wq = xq + (size_t)TOKENS * KDIM;
    float* gscale   = (float*)(wq + (size_t)NDIM * KDIM);

    bitnet_quant_x<<<TOKENS, 256, 0, stream>>>(x, scale, xq, gscale);
    bitnet_quant_w<<<(int)(((size_t)NDIM * KDIM) / 1024), 256, 0, stream>>>(w, wq);
    bitnet_gemm_iu8<<<dim3(TOKENS / TILE_M, NDIM / TILE_N), 256, 0, stream>>>(
        xq, wq, gscale, out);
}
